// PTrans2_19928648253729
// MI455X (gfx1250) — matrix operand order repaired and checked by reading, not compiled, not run on hardware
//
#include <hip/hip_runtime.h>
#include <hip/hip_bf16.h>

// Problem constants (from reference)
#define DD 4
#define CC 64
#define NN 4096          // H*W = 64*64
#define QKC 32           // q/k channels
#define TN 16            // attention rows per workgroup tile
#define S_STRIDE 4097    // padded LDS row stride (floats) to avoid bank conflicts

typedef __attribute__((ext_vector_type(16))) _Float16 v16h;
typedef __attribute__((ext_vector_type(8)))  float    v8f;

// ---------------------------------------------------------------------------
// Kernel 1: 3x3 SAME conv for q and k, writing directly into WMMA-packed f16.
//   qA layout: A-matrix 16x32 f16 tiles:  [d][tile=n/16][lane 0..31][elem 0..15]
//     element (n,c): r=n%16, h=(c>>3)&1 -> lane=r+16h, elem=(c&7)+8*(c>>4)
//   kB layout: B-matrix 32x16 f16 tiles:  [d][chunk=m/16][lane][elem]
//     element (c,m): lane=(m%16)+16*((c>>3)&1), elem=(c&7)+8*(c>>4)
// ---------------------------------------------------------------------------
__global__ __launch_bounds__(256)
void conv_pack_kernel(const float* __restrict__ x,
                      const float* __restrict__ qw, const float* __restrict__ qb,
                      const float* __restrict__ kw, const float* __restrict__ kb,
                      _Float16* __restrict__ qA, _Float16* __restrict__ kB)
{
    int idx = blockIdx.x * blockDim.x + threadIdx.x;   // D*64*N threads
    int n   = idx & (NN - 1);
    int o64 = (idx >> 12) & 63;
    int d   = idx >> 18;
    if (d >= DD) return;
    int y = n >> 6, xc = n & 63;
    bool isq = (o64 < QKC);
    int  o   = o64 & (QKC - 1);

    const float* wgt = (isq ? qw : kw) + (size_t)o * CC * 9;
    float acc = (isq ? qb : kb)[o];
    const float* xin = x + (size_t)d * CC * NN;
    for (int c = 0; c < CC; ++c) {
        const float* xch = xin + (size_t)c * NN;
        const float* wch = wgt + c * 9;
        #pragma unroll
        for (int dy = -1; dy <= 1; ++dy) {
            int yy = y + dy;
            if (yy < 0 || yy > 63) continue;
            #pragma unroll
            for (int dx = -1; dx <= 1; ++dx) {
                int xx = xc + dx;
                if (xx < 0 || xx > 63) continue;
                acc += xch[yy * 64 + xx] * wch[(dy + 1) * 3 + (dx + 1)];
            }
        }
    }
    _Float16 hv = (_Float16)acc;
    if (isq) {
        int j = n >> 4, r = n & 15;
        int h = (o >> 3) & 1;
        int lanep = r + 16 * h;
        int elem  = (o & 7) + 8 * (o >> 4);
        qA[((size_t)(d * 256 + j) * 32 + lanep) * 16 + elem] = hv;
    } else {
        int j = n >> 4;
        int lanep = (n & 15) + 16 * ((o >> 3) & 1);
        int elem  = (o & 7) + 8 * (o >> 4);
        kB[((size_t)(d * 256 + j) * 32 + lanep) * 16 + elem] = hv;
    }
}

// ---------------------------------------------------------------------------
// Kernel 2: pack v = x into B-matrix 32x16 f16 tiles for the second GEMM.
//   vB: [d][chunk=m/32][cgroup=c/16][lane][elem]
//     ml=m%32: lane=(c%16)+16*((ml>>3)&1), elem=(ml&7)+8*(ml>>4)
// ---------------------------------------------------------------------------
__global__ __launch_bounds__(256)
void pack_v_kernel(const float* __restrict__ x, _Float16* __restrict__ vB)
{
    int idx = blockIdx.x * blockDim.x + threadIdx.x;   // D*C*N threads
    int m = idx & (NN - 1);
    int c = (idx >> 12) & 63;
    int d = idx >> 18;
    if (d >= DD) return;
    float v = x[idx];
    int j  = m >> 5;
    int ml = m & 31;
    int g  = c >> 4;
    int lanep = (c & 15) + 16 * ((ml >> 3) & 1);
    int elem  = (ml & 7) + 8 * (ml >> 4);
    vB[(((size_t)(d * 128 + j) * 4 + g) * 32 + lanep) * 16 + elem] = (_Float16)v;
}

// ---------------------------------------------------------------------------
// Kernel 3: fused double-softmax attention + output GEMM + residual.
// One workgroup = one (d, 16-row tile). 512 threads = 16 wave32.
// LDS: full 16x4096 f32 score block (padded) + 16x64 out tile + row stats.
// ---------------------------------------------------------------------------
__global__ __launch_bounds__(512, 1)
void attn_kernel(const float* __restrict__ atten,   // [D][N][N]
                 const float* __restrict__ x,       // [D][C][N]
                 const _Float16* __restrict__ qA,
                 const _Float16* __restrict__ kB,
                 const _Float16* __restrict__ vB,
                 const float* __restrict__ gamma,
                 float* __restrict__ out)
{
    __shared__ float S[TN * S_STRIDE];      // 262,208 B score block
    __shared__ float outbuf[TN * CC];       // 4 KB output tile accumulator
    __shared__ float rowmax1[TN], rowinv1[TN], rowinv2[TN];

    const int d    = blockIdx.x >> 8;       // 0..3
    const int tile = blockIdx.x & 255;      // 0..255
    const int n0   = tile * TN;
    const int tid  = threadIdx.x;
    const int lane = tid & 31;
    const int w    = tid >> 5;              // wave id 0..15

    for (int i = tid; i < TN * CC; i += 512) outbuf[i] = 0.0f;

    // --- load q A-tile (16x32 f16), identical for all waves ---
    v16h a_q = *(const v16h*)(qA + ((size_t)(d * 256 + tile) * 32 + lane) * 16);

    // --- Pass A: S = q^T k via WMMA, 256 chunks of 16 columns ---
    const int nbase = 8 * (lane >> 4);
    const int mloc  = lane & 15;
    for (int jj = 0; jj < 16; ++jj) {
        int j = w + jj * 16;                // 0..255
        v16h b_k = *(const v16h*)(kB + ((size_t)(d * 256 + j) * 32 + lane) * 16);
        v8f acc = {};
        acc = __builtin_amdgcn_wmma_f32_16x16x32_f16(false, a_q, false, b_k,
                                                     (short)0, acc, false, false);
        int m = j * 16 + mloc;
        #pragma unroll
        for (int r = 0; r < 8; ++r)
            S[(nbase + r) * S_STRIDE + m] = acc[r];
    }
    __syncthreads();

    // --- Softmax-1 stats: wave w owns row w ---
    {
        float* Srow = &S[w * S_STRIDE];
        float mx = -INFINITY;
        for (int m = lane; m < NN; m += 32) mx = fmaxf(mx, Srow[m]);
        #pragma unroll
        for (int o = 16; o; o >>= 1) mx = fmaxf(mx, __shfl_xor(mx, o, 32));
        float sum = 0.0f;
        for (int m = lane; m < NN; m += 32) sum += __expf(Srow[m] - mx);
        #pragma unroll
        for (int o = 16; o; o >>= 1) sum += __shfl_xor(sum, o, 32);
        if (lane == 0) { rowmax1[w] = mx; rowinv1[w] = 1.0f / sum; }
    }
    __syncthreads();

    // --- Pass B: t = softmax1 + atten ; softmax-2 stats ; store exp(t-max2) ---
    {
        float mx1  = rowmax1[w];
        float inv1 = rowinv1[w];
        float* Srow = &S[w * S_STRIDE];
        const float* arow = atten + ((size_t)d * NN + (n0 + w)) * NN;
        float mx2 = -INFINITY;
        for (int m = lane; m < NN; m += 32) {
            float t = __expf(Srow[m] - mx1) * inv1 + arow[m];
            Srow[m] = t;
            mx2 = fmaxf(mx2, t);
        }
        #pragma unroll
        for (int o = 16; o; o >>= 1) mx2 = fmaxf(mx2, __shfl_xor(mx2, o, 32));
        float sum2 = 0.0f;
        for (int m = lane; m < NN; m += 32) {
            float e = __expf(Srow[m] - mx2);
            Srow[m] = e;                    // keep unnormalized exp
            sum2 += e;
        }
        #pragma unroll
        for (int o = 16; o; o >>= 1) sum2 += __shfl_xor(sum2, o, 32);
        if (lane == 0) rowinv2[w] = 1.0f / sum2;
    }
    __syncthreads();

    // --- Pass C: out(16n x 64c) = E (16xN) * v^T (N x 64) via WMMA ---
    v8f acc0 = {}, acc1 = {}, acc2 = {}, acc3 = {};
    {
        const int nrow = lane & 15;
        const int h    = lane >> 4;
        const float* Srow = &S[nrow * S_STRIDE];
        for (int jj = 0; jj < 8; ++jj) {
            int j  = w + jj * 16;           // 0..127 ; m0 = 32*j
            int m0 = j * 32;
            v16h aE;
            #pragma unroll
            for (int e = 0; e < 8; ++e)
                aE[e] = (_Float16)Srow[m0 + 8 * h + e];
            #pragma unroll
            for (int e = 0; e < 8; ++e)
                aE[8 + e] = (_Float16)Srow[m0 + 16 + 8 * h + e];
            const _Float16* vb = vB + (((size_t)(d * 128 + j) * 4) * 32 + lane) * 16;
            v16h b0 = *(const v16h*)(vb);
            v16h b1 = *(const v16h*)(vb + 32 * 16);
            v16h b2 = *(const v16h*)(vb + 2 * 32 * 16);
            v16h b3 = *(const v16h*)(vb + 3 * 32 * 16);
            acc0 = __builtin_amdgcn_wmma_f32_16x16x32_f16(false, aE, false, b0, (short)0, acc0, false, false);
            acc1 = __builtin_amdgcn_wmma_f32_16x16x32_f16(false, aE, false, b1, (short)0, acc1, false, false);
            acc2 = __builtin_amdgcn_wmma_f32_16x16x32_f16(false, aE, false, b2, (short)0, acc2, false, false);
            acc3 = __builtin_amdgcn_wmma_f32_16x16x32_f16(false, aE, false, b3, (short)0, acc3, false, false);
        }
    }
    // cross-wave reduce into outbuf (LDS float atomics)
    {
        int n = 8 * (lane >> 4);
        int c = lane & 15;
        #pragma unroll
        for (int r = 0; r < 8; ++r) {
            atomicAdd(&outbuf[(n + r) * CC + c],      acc0[r]);
            atomicAdd(&outbuf[(n + r) * CC + 16 + c], acc1[r]);
            atomicAdd(&outbuf[(n + r) * CC + 32 + c], acc2[r]);
            atomicAdd(&outbuf[(n + r) * CC + 48 + c], acc3[r]);
        }
    }
    __syncthreads();

    // --- final: out = gamma * (acc * inv_sum2) + x ---
    {
        float g = gamma[0];
        for (int i = tid; i < TN * CC; i += 512) {
            int c = i >> 4;
            int n = i & 15;
            size_t gi = ((size_t)d * CC + c) * NN + n0 + n;
            out[gi] = g * outbuf[n * CC + c] * rowinv2[n] + x[gi];
        }
    }
}

// ---------------------------------------------------------------------------
// Host launcher
// inputs: 0:x 1:atten 2:q_w 3:q_b 4:k_w 5:k_b 6:gamma
// ---------------------------------------------------------------------------
extern "C" void kernel_launch(void* const* d_in, const int* in_sizes, int n_in,
                              void* d_out, int out_size, void* d_ws, size_t ws_size,
                              hipStream_t stream)
{
    const float* x     = (const float*)d_in[0];
    const float* atten = (const float*)d_in[1];
    const float* q_w   = (const float*)d_in[2];
    const float* q_b   = (const float*)d_in[3];
    const float* k_w   = (const float*)d_in[4];
    const float* k_b   = (const float*)d_in[5];
    const float* gamma = (const float*)d_in[6];
    float* out = (float*)d_out;

    // workspace carve-up (halves): qA 512K, kB 512K, vB 1M  -> 4 MB total
    _Float16* qA = (_Float16*)d_ws;
    _Float16* kB = qA + (size_t)DD * 256 * 32 * 16;
    _Float16* vB = kB + (size_t)DD * 256 * 32 * 16;

    {
        int total = DD * 64 * NN;            // 1,048,576
        conv_pack_kernel<<<total / 256, 256, 0, stream>>>(x, q_w, q_b, k_w, k_b, qA, kB);
    }
    {
        int total = DD * CC * NN;            // 1,048,576
        pack_v_kernel<<<total / 256, 256, 0, stream>>>(x, vB);
    }
    {
        attn_kernel<<<DD * (NN / TN), 512, 0, stream>>>(atten, x, qA, kB, vB, gamma, out);
    }
}
